// Sheet_1116691497263
// MI455X (gfx1250) — matrix operand order repaired and checked by reading, not compiled, not run on hardware
//
#include <hip/hip_runtime.h>
#include <hip/hip_bf16.h>
#include <math.h>

#define N_SEQ 4096
#define C_DIM 128
#define D_DIM 64
#define B_SZ  8
#define EPS   1e-5f
#define LDS_STRIDE 4100   // 4096 + 4: rotates banks between rows

typedef __attribute__((ext_vector_type(16))) __bf16 v16bf;
typedef __attribute__((ext_vector_type(8)))  float  v8f;

union BF16Frag {
    v16bf v;
    unsigned short u[16];
    uint4 q[2];
};

static __device__ __forceinline__ unsigned short f2bf(float f) {
    unsigned int u = __float_as_uint(f);
    u += 0x7FFFu + ((u >> 16) & 1u);   // round-to-nearest-even
    return (unsigned short)(u >> 16);
}

// ---------------------------------------------------------------------------
// Kernel 1: norm1 + q/k projections. One block per position n; loop all 8
// batches inside so q_w/k_w (268 MB) are streamed exactly once.
// ---------------------------------------------------------------------------
__global__ __launch_bounds__(128)
void norm_qk_kernel(const float* __restrict__ x,
                    const float* __restrict__ n1w, const float* __restrict__ n1b,
                    const float* __restrict__ qw,  const float* __restrict__ qb,
                    const float* __restrict__ kw,
                    unsigned short* __restrict__ hT,   // [B,C,N] bf16 (transposed)
                    unsigned short* __restrict__ qbf,  // [B,N,D] bf16
                    unsigned short* __restrict__ kbf)  // [B,N,D] bf16
{
    __shared__ float hs[B_SZ][C_DIM];
    const int n    = blockIdx.x;
    const int t    = threadIdx.x;
    const int wid  = t >> 5;
    const int lane = t & 31;

    // Phase A: layernorm. Wave w handles batches w and w+4 (32 lanes x 4 elems).
    for (int bb = 0; bb < 2; ++bb) {
        const int b = wid + bb * 4;
        const float* xr = x + ((size_t)b * N_SEQ + n) * C_DIM;
        float4 xv = *(const float4*)(xr + lane * 4);
        float s = xv.x + xv.y + xv.z + xv.w;
        #pragma unroll
        for (int m = 16; m >= 1; m >>= 1) s += __shfl_xor(s, m, 32);
        const float mu = s * (1.0f / C_DIM);
        const float d0 = xv.x - mu, d1 = xv.y - mu, d2 = xv.z - mu, d3 = xv.w - mu;
        float vs = d0 * d0 + d1 * d1 + d2 * d2 + d3 * d3;
        #pragma unroll
        for (int m = 16; m >= 1; m >>= 1) vs += __shfl_xor(vs, m, 32);
        const float rstd = rsqrtf(vs * (1.0f / C_DIM) + EPS);
        const float dd[4] = {d0, d1, d2, d3};
        #pragma unroll
        for (int j = 0; j < 4; ++j) {
            const int c = lane * 4 + j;
            const float h = dd[j] * rstd * n1w[(size_t)n * C_DIM + c]
                          + n1b[(size_t)n * C_DIM + c];
            hs[b][c] = h;
            hT[((size_t)b * C_DIM + c) * N_SEQ + n] = f2bf(h);
        }
    }
    __syncthreads();

    // Phase B: 128 threads = {q,k} x 64 outputs; full dot over C for 8 batches.
    const int proj = t >> 6;      // 0 = q, 1 = k
    const int d    = t & 63;
    const float* wrow = (proj ? kw : qw) + ((size_t)n * D_DIM + d) * C_DIM;
    float acc[B_SZ];
    #pragma unroll
    for (int b = 0; b < B_SZ; ++b) acc[b] = 0.0f;
    for (int c = 0; c < C_DIM; c += 4) {
        const float4 wv = *(const float4*)(wrow + c);
        #pragma unroll
        for (int b = 0; b < B_SZ; ++b)
            acc[b] += hs[b][c] * wv.x + hs[b][c + 1] * wv.y
                    + hs[b][c + 2] * wv.z + hs[b][c + 3] * wv.w;
    }
    const float scale = 0.08838834764831845f;  // C^-0.5
    #pragma unroll
    for (int b = 0; b < B_SZ; ++b) {
        if (proj == 0) {
            const float v = scale * (acc[b] + qb[(size_t)n * D_DIM + d]);
            qbf[((size_t)b * N_SEQ + n) * D_DIM + d] = f2bf(v);
        } else {
            kbf[((size_t)b * N_SEQ + n) * D_DIM + d] = f2bf(acc[b]);
        }
    }
}

// ---------------------------------------------------------------------------
// Kernel 2 (fully fused attention): S = q k^T + bias, softmax, AND
// pooled = attn @ h, one block per 16-row stripe using the 320 KB CDNA5 LDS.
//   - 8 waves x 32 column tiles x 2 bf16 WMMAs -> raw scores into LDS
//   - block-wide row max / exp / sum out of LDS
//   - normalized attn written to global exactly once (the only attn traffic)
//   - pooled tile computed from the LDS exp values (A from LDS, h from L2),
//     scaled by 1/rowsum at the end -> attn is never re-read from HBM
// ---------------------------------------------------------------------------
__global__ __launch_bounds__(256)
void fused_attn_kernel(const unsigned short* __restrict__ qbf,
                       const unsigned short* __restrict__ kbf,
                       const float* __restrict__ attn_bias,
                       const unsigned short* __restrict__ hT,
                       float* __restrict__ out_attn,
                       float* __restrict__ pooled)
{
    __shared__ float S[16 * LDS_STRIDE];   // 262,400 B score stripe
    __shared__ float rmax[16][17];
    __shared__ float rsum[16][17];

    const int t    = threadIdx.x;
    const int wid  = t >> 5;
    const int lane = t & 31;
    const int b    = blockIdx.x >> 8;      // grid = B * 256
    const int nt   = blockIdx.x & 255;
    const int n0   = nt * 16;
    const int l15  = lane & 15;
    const int hi   = lane >> 4;

    // ---- WMMA phase: A fragments (q rows) loaded once, reused for 32 tiles.
    const size_t qrow = ((size_t)b * N_SEQ + (n0 + l15)) * D_DIM;
    BF16Frag a0, a1;
    a0.q[0] = *(const uint4*)(qbf + qrow + 0  + hi * 8);
    a0.q[1] = *(const uint4*)(qbf + qrow + 16 + hi * 8);
    a1.q[0] = *(const uint4*)(qbf + qrow + 32 + hi * 8);
    a1.q[1] = *(const uint4*)(qbf + qrow + 48 + hi * 8);

    #pragma unroll 4
    for (int i = 0; i < 32; ++i) {
        const int m0 = (wid * 32 + i) * 16;
        const size_t krow = ((size_t)b * N_SEQ + (m0 + l15)) * D_DIM;
        BF16Frag b0, b1;
        b0.q[0] = *(const uint4*)(kbf + krow + hi * 8);
        b0.q[1] = *(const uint4*)(kbf + krow + 16 + hi * 8);
        b1.q[0] = *(const uint4*)(kbf + krow + 32 + hi * 8);
        b1.q[1] = *(const uint4*)(kbf + krow + 48 + hi * 8);
        v8f acc = {};
        acc = __builtin_amdgcn_wmma_f32_16x16x32_bf16(false, a0.v, false, b0.v,
                                                      (short)0, acc, false, false);
        acc = __builtin_amdgcn_wmma_f32_16x16x32_bf16(false, a1.v, false, b1.v,
                                                      (short)0, acc, false, false);
        #pragma unroll
        for (int r = 0; r < 8; ++r)
            S[(r + hi * 8) * LDS_STRIDE + m0 + l15] = acc[r];
    }
    __syncthreads();

    // ---- softmax phase: thread owns row (t>>4), strided col segments.
    const int row = t >> 4;
    const int seg = t & 15;
    const float* brow = attn_bias + (size_t)(n0 + row) * N_SEQ;
    float* srow = &S[row * LDS_STRIDE];

    // pass 1: add bias, track max
    float mx = -3.4e38f;
    for (int k = 0; k < 64; ++k) {
        const int col = seg * 4 + k * 64;
        float4 sv = *(float4*)(srow + col);
        const float4 bv = *(const float4*)(brow + col);
        sv.x += bv.x; sv.y += bv.y; sv.z += bv.z; sv.w += bv.w;
        *(float4*)(srow + col) = sv;
        mx = fmaxf(mx, fmaxf(fmaxf(sv.x, sv.y), fmaxf(sv.z, sv.w)));
    }
    rmax[row][seg] = mx;
    __syncthreads();
    if (t < 16) {
        float m = rmax[t][0];
        #pragma unroll
        for (int j = 1; j < 16; ++j) m = fmaxf(m, rmax[t][j]);
        rmax[t][16] = m;
    }
    __syncthreads();
    const float mrow = rmax[row][16];

    // pass 2: exp in place, accumulate sum
    float sum = 0.0f;
    for (int k = 0; k < 64; ++k) {
        const int col = seg * 4 + k * 64;
        float4 sv = *(float4*)(srow + col);
        sv.x = __expf(sv.x - mrow); sv.y = __expf(sv.y - mrow);
        sv.z = __expf(sv.z - mrow); sv.w = __expf(sv.w - mrow);
        *(float4*)(srow + col) = sv;
        sum += sv.x + sv.y + sv.z + sv.w;
    }
    rsum[row][seg] = sum;
    __syncthreads();
    if (t < 16) {
        float s = 0.0f;
        #pragma unroll
        for (int j = 0; j < 16; ++j) s += rsum[t][j];
        rsum[t][16] = 1.0f / s;
    }
    __syncthreads();
    const float inv = rsum[row][16];

    // pass 3: normalized attn write to global (single pass over attn output)
    {
        float* orow = out_attn + ((size_t)b * N_SEQ + n0 + row) * N_SEQ;
        for (int k = 0; k < 64; ++k) {
            const int col = seg * 4 + k * 64;
            float4 sv = *(float4*)(srow + col);
            sv.x *= inv; sv.y *= inv; sv.z *= inv; sv.w *= inv;
            *(float4*)(orow + col) = sv;
        }
    }

    // ---- pooled phase: wave wid owns output column tile c0 = wid*16.
    // A = exp values straight from LDS (f32 -> bf16), B = hT streamed from L2.
    // Row-linearity lets us apply 1/rowsum on the accumulator at the end.
    {
        const int c0 = wid * 16;
        const float* arow = &S[l15 * LDS_STRIDE];
        const unsigned short* hrow = hT + ((size_t)b * C_DIM + c0 + l15) * N_SEQ;

        v8f acc = {};
        #pragma unroll 2
        for (int k0 = 0; k0 < N_SEQ; k0 += 32) {
            const float4 s0a = *(const float4*)(arow + k0 + hi * 8);
            const float4 s0b = *(const float4*)(arow + k0 + hi * 8 + 4);
            const float4 s1a = *(const float4*)(arow + k0 + 16 + hi * 8);
            const float4 s1b = *(const float4*)(arow + k0 + 16 + hi * 8 + 4);
            BF16Frag a;
            a.u[0]  = f2bf(s0a.x); a.u[1]  = f2bf(s0a.y);
            a.u[2]  = f2bf(s0a.z); a.u[3]  = f2bf(s0a.w);
            a.u[4]  = f2bf(s0b.x); a.u[5]  = f2bf(s0b.y);
            a.u[6]  = f2bf(s0b.z); a.u[7]  = f2bf(s0b.w);
            a.u[8]  = f2bf(s1a.x); a.u[9]  = f2bf(s1a.y);
            a.u[10] = f2bf(s1a.z); a.u[11] = f2bf(s1a.w);
            a.u[12] = f2bf(s1b.x); a.u[13] = f2bf(s1b.y);
            a.u[14] = f2bf(s1b.z); a.u[15] = f2bf(s1b.w);
            BF16Frag bf;
            bf.q[0] = *(const uint4*)(hrow + k0 + hi * 8);
            bf.q[1] = *(const uint4*)(hrow + k0 + 16 + hi * 8);
            acc = __builtin_amdgcn_wmma_f32_16x16x32_bf16(
                      false, a.v, false, bf.v, (short)0, acc, false, false);
        }
        #pragma unroll
        for (int r = 0; r < 8; ++r) {
            const int prow = r + hi * 8;
            pooled[((size_t)b * N_SEQ + n0 + prow) * C_DIM + c0 + l15] =
                acc[r] * rsum[prow][16];
        }
    }
}

// ---------------------------------------------------------------------------
// Kernel 3: norm2 + fc1 + gelu + fc2 + residual. Block per n, batches looped
// inside so fc1_w/fc2_w are streamed once.
// ---------------------------------------------------------------------------
__global__ __launch_bounds__(128)
void mlp_kernel(const float* __restrict__ pooled,
                const float* __restrict__ n2w, const float* __restrict__ n2b,
                const float* __restrict__ fc1w, const float* __restrict__ fc1b,
                const float* __restrict__ fc2w,
                float* __restrict__ out_x)
{
    __shared__ float hs[B_SZ][C_DIM];
    __shared__ float hid[B_SZ][D_DIM];
    const int n    = blockIdx.x;
    const int t    = threadIdx.x;
    const int wid  = t >> 5;
    const int lane = t & 31;

    // Phase A: norm2(pooled)
    for (int bb = 0; bb < 2; ++bb) {
        const int b = wid + bb * 4;
        const float* xr = pooled + ((size_t)b * N_SEQ + n) * C_DIM;
        float4 xv = *(const float4*)(xr + lane * 4);
        float s = xv.x + xv.y + xv.z + xv.w;
        #pragma unroll
        for (int m = 16; m >= 1; m >>= 1) s += __shfl_xor(s, m, 32);
        const float mu = s * (1.0f / C_DIM);
        const float d0 = xv.x - mu, d1 = xv.y - mu, d2 = xv.z - mu, d3 = xv.w - mu;
        float vs = d0 * d0 + d1 * d1 + d2 * d2 + d3 * d3;
        #pragma unroll
        for (int m = 16; m >= 1; m >>= 1) vs += __shfl_xor(vs, m, 32);
        const float rstd = rsqrtf(vs * (1.0f / C_DIM) + EPS);
        const float dd[4] = {d0, d1, d2, d3};
        #pragma unroll
        for (int j = 0; j < 4; ++j) {
            const int c = lane * 4 + j;
            hs[b][c] = dd[j] * rstd * n2w[(size_t)n * C_DIM + c]
                     + n2b[(size_t)n * C_DIM + c];
        }
    }
    __syncthreads();

    // Phase B: hid = gelu(h2 @ fc1_w[n]^T + fc1_b[n])
    if (t < D_DIM) {
        const int d = t;
        const float* wrow = fc1w + ((size_t)n * D_DIM + d) * C_DIM;
        float acc[B_SZ];
        #pragma unroll
        for (int b = 0; b < B_SZ; ++b) acc[b] = 0.0f;
        for (int c = 0; c < C_DIM; c += 4) {
            const float4 wv = *(const float4*)(wrow + c);
            #pragma unroll
            for (int b = 0; b < B_SZ; ++b)
                acc[b] += hs[b][c] * wv.x + hs[b][c + 1] * wv.y
                        + hs[b][c + 2] * wv.z + hs[b][c + 3] * wv.w;
        }
        const float bias = fc1b[(size_t)n * D_DIM + d];
        #pragma unroll
        for (int b = 0; b < B_SZ; ++b) {
            const float u = acc[b] + bias;
            hid[b][d] = 0.5f * u * (1.0f + erff(u * 0.70710678118654752f));
        }
    }
    __syncthreads();

    // Phase C: out = pooled + hid @ fc2_w[n]^T
    {
        const int c = t;
        const float* wrow = fc2w + ((size_t)n * C_DIM + c) * D_DIM;
        float acc[B_SZ];
        #pragma unroll
        for (int b = 0; b < B_SZ; ++b) acc[b] = 0.0f;
        for (int d = 0; d < D_DIM; d += 4) {
            const float4 wv = *(const float4*)(wrow + d);
            #pragma unroll
            for (int b = 0; b < B_SZ; ++b)
                acc[b] += hid[b][d] * wv.x + hid[b][d + 1] * wv.y
                        + hid[b][d + 2] * wv.z + hid[b][d + 3] * wv.w;
        }
        #pragma unroll
        for (int b = 0; b < B_SZ; ++b) {
            const size_t idx = ((size_t)b * N_SEQ + n) * C_DIM + c;
            out_x[idx] = pooled[idx] + acc[b];
        }
    }
}

// ---------------------------------------------------------------------------
extern "C" void kernel_launch(void* const* d_in, const int* in_sizes, int n_in,
                              void* d_out, int out_size, void* d_ws, size_t ws_size,
                              hipStream_t stream)
{
    const float* x     = (const float*)d_in[0];
    const float* n1w   = (const float*)d_in[1];
    const float* n1b   = (const float*)d_in[2];
    const float* qw    = (const float*)d_in[3];
    const float* qb    = (const float*)d_in[4];
    const float* kw    = (const float*)d_in[5];
    const float* abias = (const float*)d_in[6];
    const float* n2w   = (const float*)d_in[7];
    const float* n2b   = (const float*)d_in[8];
    const float* fc1w  = (const float*)d_in[9];
    const float* fc1b  = (const float*)d_in[10];
    const float* fc2w  = (const float*)d_in[11];

    float* out_x    = (float*)d_out;                              // [B,N,C]
    float* out_attn = out_x + (size_t)B_SZ * N_SEQ * C_DIM;       // [B,N,N]

    char* ws = (char*)d_ws;
    unsigned short* hT  = (unsigned short*)(ws);                  //  8 MB [B,C,N]
    unsigned short* qbf = (unsigned short*)(ws + (8u  << 20));    //  4 MB [B,N,D]
    unsigned short* kbf = (unsigned short*)(ws + (12u << 20));    //  4 MB [B,N,D]
    float*          pooled = (float*)(ws + (16u << 20));          // 16 MB [B,N,C]

    norm_qk_kernel<<<N_SEQ, 128, 0, stream>>>(x, n1w, n1b, qw, qb, kw,
                                              hT, qbf, kbf);
    fused_attn_kernel<<<B_SZ * 256, 256, 0, stream>>>(qbf, kbf, abias, hT,
                                                      out_attn, pooled);
    mlp_kernel<<<N_SEQ, 128, 0, stream>>>(pooled, n2w, n2b, fc1w, fc1b, fc2w,
                                          out_x);
}
